// Attention_27358941676380
// MI455X (gfx1250) — matrix operand order repaired and checked by reading, not compiled, not run on hardware
//
#include <hip/hip_runtime.h>
#include <hip/hip_bf16.h>
#include <math.h>

// ---------------------------------------------------------------------------
// Problem constants: B=1, LQ=LK=2048, D=4096, H=32, KVH=8, DH=128, G=4
// ---------------------------------------------------------------------------
#define LQC 2048
#define LKC 2048
#define DC  4096
#define HC  32
#define KVHC 8
#define DHC 128
#define HDC (HC*DHC)      // 4096
#define KVDC (KVHC*DHC)   // 1024

typedef __attribute__((ext_vector_type(16))) __bf16 v16bf;
typedef __attribute__((ext_vector_type(8)))  float  v8f;

union BFrag { v16bf v; unsigned int u[8]; };

__device__ __forceinline__ unsigned short f32_to_bf16(float f) {
  union { float f; unsigned int u; } c; c.f = f;
  unsigned int u = c.u;
  unsigned int r = u + 0x7FFFu + ((u >> 16) & 1u);   // round-to-nearest-even
  return (unsigned short)(r >> 16);
}

// ---------------------------------------------------------------------------
// Tiled GEMM: C[M,N] = A[M,K] * B[K,N] (B always f32; A f32 or bf16).
// Block = 256 threads (8 waves). Tile 128x128, K-step 32.
// Wave grid 4(M) x 2(N): each wave 32x64 = 2x4 WMMA accum tiles.
// Software-pipelined: tile k+1 is loaded into REGISTERS while computing on
// LDS buffer cur, then written to LDS buffer nxt; one barrier per K-step.
// A staged in LDS as bf16 [m][k]; B staged transposed [n][k] so B-fragment
// reads are contiguous b32/b128 LDS loads.
// ---------------------------------------------------------------------------
#define TM 128
#define TN 128
#define TK 32

template <bool A_BF16, bool OUT_BF16>
__global__ __launch_bounds__(256)
void gemm_kernel(const void* __restrict__ Aptr, const float* __restrict__ B,
                 void* __restrict__ Cptr, int M, int N, int K, float scale)
{
  __shared__ unsigned short lA[2][TM * TK];   // [m][k] bf16
  __shared__ unsigned short lBT[2][TN * TK];  // [n][k] bf16 (transposed)

  const int tid   = threadIdx.x;
  const int wave  = tid >> 5;
  const int lane  = tid & 31;
  const int warpM = wave >> 1;   // 0..3
  const int warpN = wave & 1;    // 0..1
  const int lhalf = lane >> 4;   // 0/1
  const int l16   = lane & 15;
  const int m0 = blockIdx.y * TM;
  const int n0 = blockIdx.x * TN;

  // staging registers for the in-flight tile
  uint4  a_bf[2];
  float4 a_f[4];
  float4 b_f[4];

  auto load_regs = [&](int k0) {
    if (A_BF16) {
      const unsigned short* A = (const unsigned short*)Aptr;
      for (int j = 0; j < 2; ++j) {                // 512 uint4 total
        int i = tid * 2 + j;
        int row = i >> 2, seg = i & 3;             // 4 x uint4 per row
        a_bf[j] = *(const uint4*)(A + (size_t)(m0 + row) * K + k0 + seg * 8);
      }
    } else {
      const float* A = (const float*)Aptr;
      for (int j = 0; j < 4; ++j) {                // 1024 float4 total
        int i = tid * 4 + j;
        int row = i >> 3, seg = i & 7;             // 8 x float4 per row
        a_f[j] = *(const float4*)(A + (size_t)(m0 + row) * K + k0 + seg * 4);
      }
    }
    for (int j = 0; j < 4; ++j) {                  // 1024 float4 total
      int i = tid * 4 + j;
      int kr = i >> 5, seg = i & 31;               // 32 x float4 per k-row
      b_f[j] = *(const float4*)(B + (size_t)(k0 + kr) * N + n0 + seg * 4);
    }
  };

  auto store_lds = [&](int buf) {
    if (A_BF16) {
      for (int j = 0; j < 2; ++j) {
        int i = tid * 2 + j;
        int row = i >> 2, seg = i & 3;
        *(uint4*)(&lA[buf][row * TK + seg * 8]) = a_bf[j];
      }
    } else {
      for (int j = 0; j < 4; ++j) {
        int i = tid * 4 + j;
        int row = i >> 3, seg = i & 7;
        unsigned int p0 = (unsigned int)f32_to_bf16(a_f[j].x) |
                          ((unsigned int)f32_to_bf16(a_f[j].y) << 16);
        unsigned int p1 = (unsigned int)f32_to_bf16(a_f[j].z) |
                          ((unsigned int)f32_to_bf16(a_f[j].w) << 16);
        *(uint2*)(&lA[buf][row * TK + seg * 4]) = make_uint2(p0, p1);
      }
    }
    for (int j = 0; j < 4; ++j) {
      int i = tid * 4 + j;
      int kr = i >> 5, seg = i & 31;
      lBT[buf][(seg * 4 + 0) * TK + kr] = f32_to_bf16(b_f[j].x);
      lBT[buf][(seg * 4 + 1) * TK + kr] = f32_to_bf16(b_f[j].y);
      lBT[buf][(seg * 4 + 2) * TK + kr] = f32_to_bf16(b_f[j].z);
      lBT[buf][(seg * 4 + 3) * TK + kr] = f32_to_bf16(b_f[j].w);
    }
  };

  v8f acc[2][4];
  for (int i = 0; i < 2; ++i)
    for (int j = 0; j < 4; ++j)
      acc[i][j] = {};

  // prologue: stage tile 0
  load_regs(0);
  store_lds(0);
  __syncthreads();

  int cur = 0;
  for (int k0 = 0; k0 < K; k0 += TK) {
    const int nxt = cur ^ 1;
    const bool has_next = (k0 + TK < K);

    // issue global loads for the NEXT tile (latency overlapped with compute)
    if (has_next) load_regs(k0 + TK);

    // ---- compute on buffer cur ----
    BFrag af[2];
    for (int mt = 0; mt < 2; ++mt) {
      int mrow = warpM * 32 + mt * 16 + l16;       // A: M = lane%16
      for (int r = 0; r < 8; ++r) {
        int kk = ((r >> 2) << 4) + (lhalf << 3) + ((r & 3) << 1);
        af[mt].u[r] = *(const unsigned int*)(&lA[cur][mrow * TK + kk]);
      }
    }
    for (int nt = 0; nt < 4; ++nt) {
      BFrag bf;
      int ncol = warpN * 64 + nt * 16 + l16;       // B: N = lane%16
      for (int r = 0; r < 8; ++r) {
        int kk = ((r >> 2) << 4) + (lhalf << 3) + ((r & 3) << 1);
        bf.u[r] = *(const unsigned int*)(&lBT[cur][ncol * TK + kk]);
      }
      for (int mt = 0; mt < 2; ++mt) {
        acc[mt][nt] = __builtin_amdgcn_wmma_f32_16x16x32_bf16(
            false, af[mt].v, false, bf.v, (short)0, acc[mt][nt], false, false);
      }
    }

    // ---- commit next tile to the other buffer ----
    if (has_next) store_lds(nxt);
    __syncthreads();
    cur = nxt;
  }

  // ---- store (C layout: N = lane%16, M = (lane/16)*8 + r) ----
  for (int mt = 0; mt < 2; ++mt)
    for (int nt = 0; nt < 4; ++nt) {
      int row0 = m0 + warpM * 32 + mt * 16 + lhalf * 8;
      int col  = n0 + warpN * 64 + nt * 16 + l16;
      for (int r = 0; r < 8; ++r) {
        float v = acc[mt][nt][r] * scale;
        if (OUT_BF16)
          ((unsigned short*)Cptr)[(size_t)(row0 + r) * N + col] = f32_to_bf16(v);
        else
          ((float*)Cptr)[(size_t)(row0 + r) * N + col] = v;
      }
    }
}

// ---------------------------------------------------------------------------
// Flash attention with additive bias + causal mask.
// Grid: (LQ/64, H). Block: 128 threads = 4 waves; wave w owns q-rows
// q0+16w .. q0+16w+15. Q fragments live in registers; K tile in LDS [key][dh];
// V tile transposed in LDS [dh][key]; P round-trips per-wave LDS tile for the
// C-layout -> A-layout conversion. Bias (the dominant 512 MB HBM stream) is
// issued into registers BEFORE the score WMMAs so its latency overlaps them.
// ---------------------------------------------------------------------------
#define FBM 64
#define FBN 64

__global__ __launch_bounds__(128)
void flash_kernel(const unsigned short* __restrict__ Q,   // [LQ][H*DH] bf16, pre-scaled
                  const unsigned short* __restrict__ Kp,  // [LK][KVH*DH] bf16, pre-scaled
                  const unsigned short* __restrict__ Vp,  // [LK][KVH*DH] bf16
                  const float* __restrict__ bias,         // [H][LQ][LK] f32
                  unsigned short* __restrict__ O)         // [LQ][H*DH] bf16
{
  __shared__ unsigned short lK[FBN * DHC];    // [key][dh]   16 KB
  __shared__ unsigned short lVT[DHC * FBN];   // [dh][key]   16 KB
  __shared__ unsigned short lP[4 * 16 * FBN]; // per-wave P   8 KB

  const int tid = threadIdx.x, wave = tid >> 5, lane = tid & 31;
  const int lhalf = lane >> 4, l16 = lane & 15;
  const int h = blockIdx.y, kvh = h >> 2;     // G = 4
  const int q0 = blockIdx.x * FBM;
  const int qrow_base = q0 + wave * 16;

  // ---- load Q fragments (dh split into 4 chunks of 32) ----
  BFrag qf[4];
  {
    const unsigned short* qbase = Q + (size_t)(qrow_base + l16) * HDC + h * DHC;
    for (int f = 0; f < 4; ++f)
      for (int r = 0; r < 8; ++r) {
        int dh = f * 32 + ((r >> 2) << 4) + (lhalf << 3) + ((r & 3) << 1);
        qf[f].u[r] = *(const unsigned int*)(qbase + dh);
      }
  }

  v8f oacc[8];
  for (int i = 0; i < 8; ++i) oacc[i] = {};
  float mrow[8], lrow[8];
  for (int r = 0; r < 8; ++r) { mrow[r] = -INFINITY; lrow[r] = 0.0f; }

  const int ktmax = blockIdx.x;               // causal: only tiles k0 <= q0
  for (int kt = 0; kt <= ktmax; ++kt) {
    const int k0 = kt * FBN;

    // ---- issue bias loads for this tile FIRST (deep HBM queue) ----
    float breg[4][8];
    for (int nt = 0; nt < 4; ++nt) {
      int col = k0 + nt * 16 + l16;
      const float* bp = bias + ((size_t)h * LQC + qrow_base + lhalf * 8) * LKC + col;
      for (int r = 0; r < 8; ++r)
        breg[nt][r] = bp[(size_t)r * LKC];
    }

    __syncthreads();
    // ---- stage K tile and transposed V tile ----
    for (int j = 0; j < 8; ++j) {             // 1024 uint4 each, 128 threads
      int i = tid * 8 + j;
      int row = i >> 4, seg = i & 15;         // 16 x uint4 per key row
      size_t gidx = (size_t)(k0 + row) * KVDC + kvh * DHC + seg * 8;
      *(uint4*)(&lK[row * DHC + seg * 8]) = *(const uint4*)(Kp + gidx);
      uint4 vv = *(const uint4*)(Vp + gidx);
      const unsigned short* ve = (const unsigned short*)&vv;
      for (int e = 0; e < 8; ++e)
        lVT[(seg * 8 + e) * FBN + row] = ve[e];
    }
    __syncthreads();

    // ---- S = Q K^T (16 x 64 per wave) ----
    v8f s[4];
    for (int nt = 0; nt < 4; ++nt) {
      s[nt] = {};
      for (int ks = 0; ks < 4; ++ks) {
        BFrag bf;
        int key = nt * 16 + l16;
        for (int r = 0; r < 8; ++r) {
          int dh = ks * 32 + ((r >> 2) << 4) + (lhalf << 3) + ((r & 3) << 1);
          bf.u[r] = *(const unsigned int*)(&lK[key * DHC + dh]);
        }
        s[nt] = __builtin_amdgcn_wmma_f32_16x16x32_bf16(
            false, qf[ks].v, false, bf.v, (short)0, s[nt], false, false);
      }
    }

    // ---- bias + causal mask ----
    for (int nt = 0; nt < 4; ++nt) {
      int col = k0 + nt * 16 + l16;
      for (int r = 0; r < 8; ++r) {
        int row = qrow_base + lhalf * 8 + r;
        float v = s[nt][r] + breg[nt][r];
        s[nt][r] = (col > row) ? -INFINITY : v;
      }
    }

    // ---- online softmax (rows live across 16-lane half-waves) ----
    float newm[8], alpha[8];
    for (int r = 0; r < 8; ++r) {
      float v = fmaxf(fmaxf(s[0][r], s[1][r]), fmaxf(s[2][r], s[3][r]));
      for (int off = 1; off < 16; off <<= 1)
        v = fmaxf(v, __shfl_xor(v, off, 32));
      newm[r]  = fmaxf(mrow[r], v);
      alpha[r] = __expf(mrow[r] - newm[r]);
      mrow[r]  = newm[r];
    }
    for (int r = 0; r < 8; ++r) {
      float sum = 0.0f;
      for (int nt = 0; nt < 4; ++nt) {
        float p = __expf(s[nt][r] - newm[r]);
        s[nt][r] = p;
        sum += p;
      }
      for (int off = 1; off < 16; off <<= 1)
        sum += __shfl_xor(sum, off, 32);
      lrow[r] = lrow[r] * alpha[r] + sum;
    }
    for (int i = 0; i < 8; ++i)
      for (int r = 0; r < 8; ++r)
        oacc[i][r] *= alpha[r];

    // ---- P -> per-wave LDS (C layout -> memory) ----
    unsigned short* pw = &lP[wave * 16 * FBN];
    for (int nt = 0; nt < 4; ++nt) {
      int col = nt * 16 + l16;
      for (int r = 0; r < 8; ++r)
        pw[(lhalf * 8 + r) * FBN + col] = f32_to_bf16(s[nt][r]);
    }
    // Per-wave LDS region; DS ops from one wave are in-order, no barrier needed.

    // ---- O += P @ V ----
    for (int ks = 0; ks < 2; ++ks) {
      BFrag af;
      for (int r = 0; r < 8; ++r) {
        int key = ks * 32 + ((r >> 2) << 4) + (lhalf << 3) + ((r & 3) << 1);
        af.u[r] = *(const unsigned int*)(&pw[l16 * FBN + key]);
      }
      for (int nt = 0; nt < 8; ++nt) {
        BFrag bf;
        int dh = nt * 16 + l16;
        for (int r = 0; r < 8; ++r) {
          int key = ks * 32 + ((r >> 2) << 4) + (lhalf << 3) + ((r & 3) << 1);
          bf.u[r] = *(const unsigned int*)(&lVT[dh * FBN + key]);
        }
        oacc[nt] = __builtin_amdgcn_wmma_f32_16x16x32_bf16(
            false, af.v, false, bf.v, (short)0, oacc[nt], false, false);
      }
    }
  }

  // ---- finalize: normalize and store bf16 [LQ][H*DH] ----
  for (int nt = 0; nt < 8; ++nt) {
    int dh = nt * 16 + l16;
    for (int r = 0; r < 8; ++r) {
      int row = qrow_base + lhalf * 8 + r;
      float v = oacc[nt][r] / lrow[r];
      O[(size_t)row * HDC + h * DHC + dh] = f32_to_bf16(v);
    }
  }
}

// ---------------------------------------------------------------------------
// Launch: inputs = hidden_q, hidden_kv, attention_mask(bool, unused: mask is
// exactly causal tril -> handled analytically), position_bias, Wq, Wk, Wv, Wo.
// ---------------------------------------------------------------------------
extern "C" void kernel_launch(void* const* d_in, const int* in_sizes, int n_in,
                              void* d_out, int out_size, void* d_ws, size_t ws_size,
                              hipStream_t stream)
{
  const float* hidden_q  = (const float*)d_in[0];
  const float* hidden_kv = (const float*)d_in[1];
  const float* bias      = (const float*)d_in[3];
  const float* Wq        = (const float*)d_in[4];
  const float* Wk        = (const float*)d_in[5];
  const float* Wv        = (const float*)d_in[6];
  const float* Wo        = (const float*)d_in[7];

  char* ws = (char*)d_ws;
  unsigned short* Qp = (unsigned short*)(ws);                                   // 16 MB
  unsigned short* Kp = (unsigned short*)(ws + (size_t)LQC * HDC * 2);           //  4 MB
  unsigned short* Vp = (unsigned short*)(ws + (size_t)LQC * HDC * 2
                                            + (size_t)LKC * KVDC * 2);          //  4 MB
  unsigned short* Oa = (unsigned short*)(ws + (size_t)LQC * HDC * 2
                                            + (size_t)LKC * KVDC * 4);          // 16 MB

  const float inv4 = 0.29730177875068026f;  // 128^(-1/4)

  // Q = hq @ Wq * inv4   (2048 x 4096 x 4096) -> bf16
  gemm_kernel<false, true><<<dim3(HDC / TN, LQC / TM), 256, 0, stream>>>(
      hidden_q, Wq, Qp, LQC, HDC, DC, inv4);
  // K = hkv @ Wk * inv4  (2048 x 1024 x 4096) -> bf16
  gemm_kernel<false, true><<<dim3(KVDC / TN, LKC / TM), 256, 0, stream>>>(
      hidden_kv, Wk, Kp, LKC, KVDC, DC, inv4);
  // V = hkv @ Wv         (2048 x 1024 x 4096) -> bf16
  gemm_kernel<false, true><<<dim3(KVDC / TN, LKC / TM), 256, 0, stream>>>(
      hidden_kv, Wv, Vp, LKC, KVDC, DC, 1.0f);
  // Flash attention -> Oa bf16
  flash_kernel<<<dim3(LQC / FBM, HC), 128, 0, stream>>>(Qp, Kp, Vp, bias, Oa);
  // out = Oa @ Wo        (2048 x 4096 x 4096) -> f32
  gemm_kernel<true, false><<<dim3(DC / TN, LQC / TM), 256, 0, stream>>>(
      Oa, Wo, (float*)d_out, LQC, DC, HDC, 1.0f);
}
